// HeteroGNN_37812892074513
// MI455X (gfx1250) — matrix operand order repaired and checked by reading, not compiled, not run on hardware
//
#include <hip/hip_runtime.h>

typedef __attribute__((ext_vector_type(16))) __bf16 v16bf;
typedef __attribute__((ext_vector_type(8)))  float  v8f;

#define NFACT 100000
#define NCOMP 10000
#define NEDGE 800000
#define DFACT 768
#define DCOMP 128
#define HID   128
#define NB    256

// ---------- helpers ----------
__device__ __forceinline__ void atomAddF(float* p, float v) {
  __hip_atomic_fetch_add(p, v, __ATOMIC_RELAXED, __HIP_MEMORY_SCOPE_AGENT);
}

// ---------- elementwise kernels ----------
__global__ void k_zero(float* p, int n) {
  int i = blockIdx.x * blockDim.x + threadIdx.x;
  if (i < n) p[i] = 0.f;
}

__global__ void k_gate_deg(const float* __restrict__ ea, const int* __restrict__ src,
                           const int* __restrict__ dst, const float* __restrict__ Wm,
                           const float* __restrict__ bm, float* __restrict__ w,
                           float* __restrict__ deg_s, float* __restrict__ deg_d, int n) {
  int e = blockIdx.x * blockDim.x + threadIdx.x;
  if (e >= n) return;
  float z = ea[2 * e] * Wm[0] + ea[2 * e + 1] * Wm[1] + bm[0];
  float g = 1.f / (1.f + __expf(-z));
  w[e] = g;
  atomAddF(&deg_s[src[e]], g);
  atomAddF(&deg_d[dst[e]], g);
}

__global__ void k_rsqrt(float* p, int n) {
  int i = blockIdx.x * blockDim.x + threadIdx.x;
  if (i < n) { float d = p[i]; p[i] = (d > 0.f) ? rsqrtf(d) : 0.f; }
}

__global__ void k_bias_relu(float* __restrict__ x, const float* __restrict__ b, int total) {
  int i = blockIdx.x * blockDim.x + threadIdx.x;
  if (i >= total) return;
  float v = x[i] + b[i & (HID - 1)];
  x[i] = v > 0.f ? v : 0.f;
}

// ---------- W pre-swizzle into WMMA B-fragment order ----------
// Output layout: [s = k/32][tn = 0..7][lane = 0..31][e = 0..15] (contiguous bf16)
// Fragment semantics (ISA 7.12.2, B 32x16): lane group g = lane>>4 holds
// K = g*8 + e (e<8), 16 + g*8 + (e-8) (e>=8), column N = tn*16 + (lane&15).
__global__ void k_swizzleW(const float* __restrict__ W, __bf16* __restrict__ y, int K) {
  int i = blockIdx.x * blockDim.x + threadIdx.x;
  if (i >= K * HID) return;
  int e    = i & 15;
  int lane = (i >> 4) & 31;
  int tn   = (i >> 9) & 7;
  int s    = i >> 12;
  int k    = s * 32 + (lane >> 4) * 8 + (e & 7) + (e >> 3) * 16;
  int col  = tn * 16 + (lane & 15);
  y[i] = static_cast<__bf16>(W[k * HID + col]);
}

// ---------- bf16 WMMA GEMM: C[M x 128] = A[M x K] * W[K x 128] ----------
// Block = 8 waves = 128x128 output tile; wave w owns rows [w*16, w*16+16).
// B slabs (32x128 = 8KB, pre-swizzled) double-buffered through LDS.
__global__ __launch_bounds__(256)
void k_gemm_bf16(const float* __restrict__ A, const __bf16* __restrict__ Bsw,
                 float* __restrict__ C, int M, int K) {
  __shared__ __align__(32) __bf16 lds[2][8 * 32 * 16];  // 2 x 8KB
  const int lane = threadIdx.x & 31;
  const int wv   = threadIdx.x >> 5;
  const int lm   = lane & 15;
  const int kh   = lane >> 4;
  int row = blockIdx.x * 128 + wv * 16 + lm;
  const int rowLd = row < M ? row : (M - 1);          // clamp: EXEC stays all-ones
  const float* Arow = A + (size_t)rowLd * K;
  const int nk = K / 32;
  v8f acc[8] = {};

  auto loadB = [&](int s, int buf) {
    const uint4* g = (const uint4*)(Bsw + (size_t)s * 4096) + threadIdx.x * 2;
    uint4* l = (uint4*)(&lds[buf][0]) + threadIdx.x * 2;
    l[0] = g[0];
    l[1] = g[1];
  };
  loadB(0, 0);

  for (int s = 0; s < nk; ++s) {
    __syncthreads();                       // slab s resident; prev slab free
    if (s + 1 < nk) loadB(s + 1, (s + 1) & 1);

    // A fragment (ISA 7.12.2, 16-bit A 16x32): lane group kh holds
    // K = {kh*8 .. kh*8+7} and {16+kh*8 .. 16+kh*8+7} of row lm.
    const float* ap = Arow + s * 32 + kh * 8;
    if (s + 2 < nk) __builtin_prefetch(ap + 64, 0, 1);   // global_prefetch A slab s+2 (two ahead)
    float4 a0 = *(const float4*)(ap + 0);
    float4 a1 = *(const float4*)(ap + 4);
    float4 a2 = *(const float4*)(ap + 16);
    float4 a3 = *(const float4*)(ap + 20);
    v16bf af;
    af[0]  = static_cast<__bf16>(a0.x);  af[1]  = static_cast<__bf16>(a0.y);
    af[2]  = static_cast<__bf16>(a0.z);  af[3]  = static_cast<__bf16>(a0.w);
    af[4]  = static_cast<__bf16>(a1.x);  af[5]  = static_cast<__bf16>(a1.y);
    af[6]  = static_cast<__bf16>(a1.z);  af[7]  = static_cast<__bf16>(a1.w);
    af[8]  = static_cast<__bf16>(a2.x);  af[9]  = static_cast<__bf16>(a2.y);
    af[10] = static_cast<__bf16>(a2.z);  af[11] = static_cast<__bf16>(a2.w);
    af[12] = static_cast<__bf16>(a3.x);  af[13] = static_cast<__bf16>(a3.y);
    af[14] = static_cast<__bf16>(a3.z);  af[15] = static_cast<__bf16>(a3.w);

    const int buf = s & 1;
    // Load all 8 B fragments into distinct registers first so the compiler
    // can clause the 16 ds_load_b128s and overlap LDS latency with WMMAs.
    v16bf bf[8];
#pragma unroll
    for (int tn = 0; tn < 8; ++tn)
      bf[tn] = *(const v16bf*)(&lds[buf][(tn * 32 + lane) * 16]);
#pragma unroll
    for (int tn = 0; tn < 8; ++tn)
      acc[tn] = __builtin_amdgcn_wmma_f32_16x16x32_bf16(false, af, false, bf[tn],
                                                        (short)0, acc[tn], false, false);
  }

  // C f32 16x16 layout: VGPR r -> row r + 8*kh, col = lane&15
#pragma unroll
  for (int tn = 0; tn < 8; ++tn) {
    const int col = tn * 16 + lm;
#pragma unroll
    for (int r = 0; r < 8; ++r) {
      int orow = blockIdx.x * 128 + wv * 16 + r + 8 * kh;
      if (orow < M) C[(size_t)orow * HID + col] = acc[tn][r];
    }
  }
}

// ---------- weighted scatter-add: out[dst] += norm * h[src], one wave/edge ----------
__global__ __launch_bounds__(256)
void k_scatter(const float* __restrict__ h, const int* __restrict__ src,
               const int* __restrict__ dst, const float* __restrict__ w,
               const float* __restrict__ dinv_s, const float* __restrict__ dinv_d,
               float* __restrict__ out, int n_edges) {
  int wid = (blockIdx.x * blockDim.x + threadIdx.x) >> 5;
  int lane = threadIdx.x & 31;
  if (wid >= n_edges) return;
  int s = src[wid], d = dst[wid];
  float nrm = w[wid] * dinv_s[s] * dinv_d[d];
  float4 hv = *(const float4*)(h + (size_t)s * HID + lane * 4);
  float* op = out + (size_t)d * HID + lane * 4;
  atomAddF(op + 0, nrm * hv.x);
  atomAddF(op + 1, nrm * hv.y);
  atomAddF(op + 2, nrm * hv.z);
  atomAddF(op + 3, nrm * hv.w);
}

// ---------- batch pooling: one wave per fact node ----------
__global__ __launch_bounds__(256)
void k_pool(const float* __restrict__ f2, const int* __restrict__ batch,
            float* __restrict__ sums, float* __restrict__ cnts) {
  int wid = (blockIdx.x * blockDim.x + threadIdx.x) >> 5;
  int lane = threadIdx.x & 31;
  if (wid >= NFACT) return;
  int b = batch[wid];
  float4 v = *(const float4*)(f2 + (size_t)wid * HID + lane * 4);
  float* op = sums + (size_t)b * HID + lane * 4;
  atomAddF(op + 0, v.x);
  atomAddF(op + 1, v.y);
  atomAddF(op + 2, v.z);
  atomAddF(op + 3, v.w);
  if (lane == 0) atomAddF(&cnts[b], 1.0f);
}

__global__ void k_classify(const float* __restrict__ sums, const float* __restrict__ cnts,
                           const float* __restrict__ Wc, const float* __restrict__ bc,
                           float* __restrict__ out) {
  int b = threadIdx.x;
  if (b >= NB) return;
  float c = cnts[b]; c = c > 1.f ? c : 1.f;
  float inv = 1.f / c;
  float acc = 0.f;
  for (int h = 0; h < HID; ++h) acc += sums[b * HID + h] * inv * Wc[h];
  out[b] = acc + bc[0];
}

// ---------- launcher ----------
extern "C" void kernel_launch(void* const* d_in, const int* in_sizes, int n_in,
                              void* d_out, int out_size, void* d_ws, size_t ws_size,
                              hipStream_t stream) {
  const float* x_fact   = (const float*)d_in[0];
  // d_in[1] (x_company), W1_cf/b1_cf, W2_fc/b2_fc feed only dead branches (f1, c2)
  const float* ea_fc    = (const float*)d_in[2];
  const float* ea_cf    = (const float*)d_in[3];
  const int*   fc_src   = (const int*)d_in[4];
  const int*   fc_dst   = (const int*)d_in[5];
  const int*   cf_src   = (const int*)d_in[6];
  const int*   cf_dst   = (const int*)d_in[7];
  const int*   batchf   = (const int*)d_in[8];
  const float* Wm       = (const float*)d_in[9];
  const float* bm       = (const float*)d_in[10];
  const float* W1_fc    = (const float*)d_in[11];
  const float* b1_fc    = (const float*)d_in[12];
  const float* W2_cf    = (const float*)d_in[17];
  const float* b2_cf    = (const float*)d_in[18];
  const float* Wc       = (const float*)d_in[19];
  const float* bc       = (const float*)d_in[20];
  float* out = (float*)d_out;

  // carve workspace (~120 MB)
  char* ws = (char*)d_ws;
  size_t off = 0;
  auto take = [&](size_t bytes) -> void* {
    void* p = ws + off;
    off = (off + bytes + 255) & ~(size_t)255;
    return p;
  };
  float*  w_fc   = (float*)take((size_t)NEDGE * 4);
  float*  w_cf   = (float*)take((size_t)NEDGE * 4);
  float*  di_fs  = (float*)take((size_t)NFACT * 4);   // rsqrt deg over fc_src (facts)
  float*  di_fd  = (float*)take((size_t)NCOMP * 4);   // rsqrt deg over fc_dst (companies)
  float*  di_cs  = (float*)take((size_t)NCOMP * 4);   // rsqrt deg over cf_src (companies)
  float*  di_cd  = (float*)take((size_t)NFACT * 4);   // rsqrt deg over cf_dst (facts)
  __bf16* W1b    = (__bf16*)take((size_t)DFACT * HID * 2);  // pre-swizzled
  __bf16* W2b    = (__bf16*)take((size_t)HID * HID * 2);    // pre-swizzled
  float*  hF     = (float*)take((size_t)NFACT * HID * 4);   // x_fact @ W1_fc
  float*  c1     = (float*)take((size_t)NCOMP * HID * 4);
  float*  hC2    = (float*)take((size_t)NCOMP * HID * 4);   // c1 @ W2_cf
  float*  f2     = (float*)take((size_t)NFACT * HID * 4);
  float*  sums   = (float*)take((size_t)NB * HID * 4);
  float*  cnts   = (float*)take((size_t)NB * 4);
  (void)ws_size; (void)n_in; (void)in_sizes; (void)out_size;

  const int T = 256;
  auto blks = [](long long n, int t) { return (int)((n + t - 1) / t); };

  // zero accumulators
  k_zero<<<blks(NFACT, T), T, 0, stream>>>(di_fs, NFACT);
  k_zero<<<blks(NCOMP, T), T, 0, stream>>>(di_fd, NCOMP);
  k_zero<<<blks(NCOMP, T), T, 0, stream>>>(di_cs, NCOMP);
  k_zero<<<blks(NFACT, T), T, 0, stream>>>(di_cd, NFACT);
  k_zero<<<blks((long long)NCOMP * HID, T), T, 0, stream>>>(c1, NCOMP * HID);
  k_zero<<<blks((long long)NFACT * HID, T), T, 0, stream>>>(f2, NFACT * HID);
  k_zero<<<blks(NB * HID, T), T, 0, stream>>>(sums, NB * HID);
  k_zero<<<blks(NB, T), T, 0, stream>>>(cnts, NB);

  // edge gates + degrees (both relations)
  k_gate_deg<<<blks(NEDGE, T), T, 0, stream>>>(ea_fc, fc_src, fc_dst, Wm, bm,
                                               w_fc, di_fs, di_fd, NEDGE);
  k_gate_deg<<<blks(NEDGE, T), T, 0, stream>>>(ea_cf, cf_src, cf_dst, Wm, bm,
                                               w_cf, di_cs, di_cd, NEDGE);
  k_rsqrt<<<blks(NFACT, T), T, 0, stream>>>(di_fs, NFACT);
  k_rsqrt<<<blks(NCOMP, T), T, 0, stream>>>(di_fd, NCOMP);
  k_rsqrt<<<blks(NCOMP, T), T, 0, stream>>>(di_cs, NCOMP);
  k_rsqrt<<<blks(NFACT, T), T, 0, stream>>>(di_cd, NFACT);

  // weights -> bf16 fragment-swizzled
  k_swizzleW<<<blks(DFACT * HID, T), T, 0, stream>>>(W1_fc, W1b, DFACT);
  k_swizzleW<<<blks(HID * HID, T), T, 0, stream>>>(W2_cf, W2b, HID);

  // layer 1 (fc relation): hF = x_fact @ W1_fc ; scatter to companies ; +bias, relu
  k_gemm_bf16<<<blks(NFACT, 128), T, 0, stream>>>(x_fact, W1b, hF, NFACT, DFACT);
  k_scatter<<<blks((long long)NEDGE * 32, T), T, 0, stream>>>(hF, fc_src, fc_dst, w_fc,
                                                              di_fs, di_fd, c1, NEDGE);
  k_bias_relu<<<blks((long long)NCOMP * HID, T), T, 0, stream>>>(c1, b1_fc, NCOMP * HID);

  // layer 2 (cf relation): hC2 = c1 @ W2_cf ; scatter to facts ; +bias, relu
  // (c2 / f1 are dead in the reference and skipped)
  k_gemm_bf16<<<blks(NCOMP, 128), T, 0, stream>>>(c1, W2b, hC2, NCOMP, HID);
  k_scatter<<<blks((long long)NEDGE * 32, T), T, 0, stream>>>(hC2, cf_src, cf_dst, w_cf,
                                                              di_cs, di_cd, f2, NEDGE);
  k_bias_relu<<<blks((long long)NFACT * HID, T), T, 0, stream>>>(f2, b2_cf, NFACT * HID);

  // readout
  k_pool<<<blks((long long)NFACT * 32, T), T, 0, stream>>>(f2, batchf, sums, cnts);
  k_classify<<<1, NB, 0, stream>>>(sums, cnts, Wc, bc, out);
}
